// ARConv_16192026706056
// MI455X (gfx1250) — matrix operand order repaired and checked by reading, not compiled, not run on hardware
//
#include <hip/hip_runtime.h>
#include <math.h>

typedef __bf16 bf16;
typedef __attribute__((ext_vector_type(16))) __bf16 v16bf;
typedef __attribute__((ext_vector_type(8)))  __bf16 v8bf;
typedef __attribute__((ext_vector_type(8)))  float  v8f;

#define BN_INV 0.99999500003749968f   // 1/sqrt(1+1e-5)

// ---------------------------------------------------------------------------
// Convert x (NCHW f32) -> NHWC f32, NHWC bf16, NHWC bf16 * 100
// ---------------------------------------------------------------------------
__global__ void prep_x_kernel(const float* __restrict__ x,
                              float* __restrict__ x_f32,
                              bf16* __restrict__ x_bf,
                              bf16* __restrict__ x100_bf)
{
    int idx = blockIdx.x * blockDim.x + threadIdx.x;      // NHWC dest index
    if (idx >= 8 * 64 * 64 * 128) return;
    int c = idx & 127;
    int w = (idx >> 7) & 63;
    int h = (idx >> 13) & 63;
    int b = idx >> 19;
    float v = x[(((size_t)b * 128 + c) * 64 + h) * 64 + w];
    x_f32[idx]   = v;
    x_bf[idx]    = (bf16)v;
    x100_bf[idx] = (bf16)(v * 100.0f);
}

// ---------------------------------------------------------------------------
// Pack OIHW f32 weights (128,128,3,3) into per-lane WMMA B-fragment layout:
// bp[((ntile*36 + kstep)*32 + lane)*16 + e]
//   = W[k = kstep*32 + (lane>>4)*16 + e][cout = ntile*16 + (lane&15)]
// with K ordering k = (ky*3+kx)*128 + cin
// ---------------------------------------------------------------------------
__global__ void pack_w_kernel(const float* __restrict__ w, bf16* __restrict__ bp)
{
    int idx = blockIdx.x * blockDim.x + threadIdx.x;
    if (idx >= 8 * 36 * 32 * 16) return;
    int e     = idx & 15;
    int lane  = (idx >> 4) & 31;
    int kstep = (idx >> 9) % 36;
    int ntile = idx / (36 * 512);
    int k    = kstep * 32 + ((lane >> 4) * 16) + e;
    int n    = lane & 15;
    int cout = ntile * 16 + n;
    int tap = k >> 7;          // k / 128
    int cin = k & 127;
    int ky = tap / 3, kx = tap - ky * 3;
    float v = w[(((size_t)cout * 128 + cin) * 3 + ky) * 3 + kx];
    bp[idx] = (bf16)v;
}

// ---------------------------------------------------------------------------
// Implicit-GEMM 3x3 conv, Cin=Cout=128, output fixed 8x64x64, via
// v_wmma_f32_16x16x32_bf16 with LDS row staging.
// Block = 256 threads = 8 waves = one output row (b,h), w = 0..63.
// Input rows (3 per block, shared by the kx taps) are cooperatively staged
// in LDS with a 136-element pixel stride (68 dwords -> conflict-free banks).
// Each wave: one 16-wide Cout tile x 4 M-subtiles; every B fragment feeds
// 4 WMMA. Fully unrolled: 3 ky x 3 kx x 4 ksteps x 4 subtiles = 144 wmma.
// ---------------------------------------------------------------------------
template <int STRIDE, int PAD, int HIN, int WIN, int ACT, bool BN,
          bool OF32, bool OBF16>
__global__ __launch_bounds__(256) void conv_wmma_t(
    const bf16* __restrict__ in,      // NHWC [8][HIN][WIN][128]
    const bf16* __restrict__ bpack,   // [8][36][32][16]
    const float* __restrict__ bias,   // [128]
    const float* __restrict__ gamma,  // BN scale (used iff BN)
    const float* __restrict__ beta,   // BN shift (used iff BN)
    float* __restrict__ out_f32,      // NHWC (used iff OF32)
    bf16*  __restrict__ out_bf16)     // NHWC (used iff OBF16)
{
    constexpr int LP = 136;                 // padded LDS pixel stride (bf16 elems)
    __shared__ bf16 rowbuf[WIN * LP];       // 17.4 KB (WIN=64) / 52.2 KB (WIN=192)

    const int tid   = threadIdx.x;
    const int lane  = tid & 31;
    const int ntile = tid >> 5;             // 0..7
    const int pix0  = blockIdx.x * 64;      // 64 pixels = one (b,h) row
    const int ho = (pix0 >> 6) & 63;
    const int bi = pix0 >> 12;
    const int wlane  = lane & 15;           // w within subtile
    const int kbaseA = (lane >> 4) * 16;

    v8f acc0 = {}, acc1 = {}, acc2 = {}, acc3 = {};
    const bf16* bp = bpack + (size_t)ntile * 36 * 512 + (size_t)lane * 16;

#pragma unroll
    for (int ky = 0; ky < 3; ++ky) {
        const int hi = ho * STRIDE + ky - PAD;          // wave-uniform
        const bool hok = (unsigned)hi < (unsigned)HIN;
        __syncthreads();                                 // rowbuf safe to overwrite
        // Cooperative row fill: WIN*4 chunks of 32 bf16 (64B each)
        {
            const bf16* src = in + ((size_t)bi * HIN + hi) * WIN * 128;
#pragma unroll
            for (int it = 0; it < (WIN * 4) / 256; ++it) {
                int pp   = tid + it * 256;
                int p    = pp >> 2;
                int part = (pp & 3) * 32;
                v8bf* dst = (v8bf*)(rowbuf + p * LP + part);
                if (hok) {
                    const v8bf* s8 = (const v8bf*)(src + p * 128 + part);
                    dst[0] = s8[0]; dst[1] = s8[1]; dst[2] = s8[2]; dst[3] = s8[3];
                } else {
                    v8bf z = {};
                    dst[0] = z; dst[1] = z; dst[2] = z; dst[3] = z;
                }
            }
        }
        __syncthreads();                                 // row visible to all waves
#pragma unroll
        for (int kx = 0; kx < 3; ++kx) {
            const int tap = ky * 3 + kx;
#pragma unroll
            for (int kq = 0; kq < 4; ++kq) {
                const int kstep = tap * 4 + kq;
                // B fragment: 32 contiguous bytes per lane (reused by 4 subtiles)
                const v16bf bfrag = *(const v16bf*)(bp + (size_t)kstep * 512);
#pragma unroll
                for (int s = 0; s < 4; ++s) {
                    const int wo = s * 16 + wlane;
                    const int wi = wo * STRIDE + kx - PAD;
                    v16bf a;
                    if ((unsigned)wi < (unsigned)WIN) {
                        const bf16* ap = rowbuf + wi * LP + kq * 32 + kbaseA;
                        v8bf a0 = *(const v8bf*)ap;         // K_local kbaseA+0..7
                        v8bf a1 = *(const v8bf*)(ap + 8);   // K_local kbaseA+8..15
                        a = __builtin_shufflevector(a0, a1, 0, 1, 2, 3, 4, 5, 6, 7,
                                                    8, 9, 10, 11, 12, 13, 14, 15);
                    } else {
#pragma unroll
                        for (int i = 0; i < 16; ++i) a[i] = (bf16)0.0f;
                    }
                    v8f& acc = (s == 0) ? acc0 : (s == 1) ? acc1 : (s == 2) ? acc2 : acc3;
                    acc = __builtin_amdgcn_wmma_f32_16x16x32_bf16(
                        false, a, false, bfrag, (short)0, acc, false, false);
                }
            }
        }
    }

    // Epilogue: C/D layout -> lane = N, VGPR r = row M (+8 for lanes 16..31)
    const int cout = ntile * 16 + (lane & 15);
    const float bs = bias[cout];
    float sc = 1.0f, sh = 0.0f;
    if (BN) { sc = gamma[cout] * BN_INV; sh = beta[cout]; }
    const int wofs = (lane >> 4) * 8;                // row-M offset within subtile
    const size_t obase = (((size_t)bi * 64 + ho) * 64) * 128 + cout;
#pragma unroll
    for (int s = 0; s < 4; ++s) {
        const v8f& acc = (s == 0) ? acc0 : (s == 1) ? acc1 : (s == 2) ? acc2 : acc3;
#pragma unroll
        for (int r = 0; r < 8; ++r) {
            int w2 = s * 16 + wofs + r;
            float v = (acc[r] + bs) * sc + sh;
            if (ACT == 1)      v = v > 0.0f ? v : 0.01f * v;
            else if (ACT == 2) v = tanhf(v);
            size_t oi = obase + (size_t)w2 * 128;
            if (OF32)  out_f32[oi]  = v;
            if (OBF16) out_bf16[oi] = (bf16)v;
        }
    }
}

// ---------------------------------------------------------------------------
// Cout=1 branch: conv3x3(128->1,pad1) + BN + lrelu, then 1x1 conv + BN +
// sigmoid scaled into [1, hw1]. One thread per output pixel.
// ---------------------------------------------------------------------------
__global__ void small_branch_kernel(
    const float* __restrict__ off,   // NHWC f32 [8][64][64][128]
    const float* __restrict__ w1, const float* __restrict__ b1,
    const float* __restrict__ g1, const float* __restrict__ be1,
    const float* __restrict__ w2, const float* __restrict__ b2,
    const float* __restrict__ g2, const float* __restrict__ be2,
    const int* __restrict__ hw_range,
    float* __restrict__ outmap)      // [8*64*64]
{
    int idx = blockIdx.x * blockDim.x + threadIdx.x;
    if (idx >= 8 * 64 * 64) return;
    int w = idx & 63, h = (idx >> 6) & 63, b = idx >> 12;
    float acc = 0.0f;
    for (int ky = 0; ky < 3; ++ky) {
        int hi = h + ky - 1;
        if ((unsigned)hi >= 64u) continue;
        for (int kx = 0; kx < 3; ++kx) {
            int wi = w + kx - 1;
            if ((unsigned)wi >= 64u) continue;
            const float* ip = off + (((size_t)b * 64 + hi) * 64 + wi) * 128;
            int t = ky * 3 + kx;
#pragma unroll 4
            for (int c = 0; c < 128; ++c) acc += ip[c] * w1[c * 9 + t];
        }
    }
    float t = (acc + b1[0]) * (g1[0] * BN_INV) + be1[0];
    t = t > 0.0f ? t : 0.01f * t;
    float u = (t * w2[0] + b2[0]) * (g2[0] * BN_INV) + be2[0];
    float s = 1.0f / (1.0f + expf(-u));
    float hw1 = (float)hw_range[1];
    outmap[idx] = s * (hw1 - 1.0f) + 1.0f;
}

// ---------------------------------------------------------------------------
// Deformable bilinear sampling -> xo NHWC bf16 [8][192][192][128]
// (padded-image semantics: coords in [0,65], row c maps to x[c-1] if 1<=c<=64)
// ---------------------------------------------------------------------------
__global__ void sample_kernel(const float* __restrict__ x,   // NHWC f32
                              const float* __restrict__ lmap,
                              const float* __restrict__ wmap,
                              bf16* __restrict__ xo)
{
    int idx = blockIdx.x * blockDim.x + threadIdx.x;
    if (idx >= 8 * 64 * 64 * 9) return;
    int n = idx % 9;
    int pix = idx / 9;
    int w = pix & 63, h = (pix >> 6) & 63, b = pix >> 12;
    int ix = n / 3, iy = n - ix * 3;

    float l  = lmap[pix];
    float wd = wmap[pix];
    float px = (float)(h + 1) + (l * (1.0f / 3.0f)) * (float)(ix - 1);
    float py = (float)(w + 1) + (wd * (1.0f / 3.0f)) * (float)(iy - 1);
    float fx = floorf(px), fy = floorf(py);
    float qltx = fminf(fmaxf(fx, 0.0f), 65.0f);
    float qlty = fminf(fmaxf(fy, 0.0f), 65.0f);
    float qrbx = fminf(fmaxf(fx + 1.0f, 0.0f), 65.0f);
    float qrby = fminf(fmaxf(fy + 1.0f, 0.0f), 65.0f);
    float pxc  = fminf(fmaxf(px, 0.0f), 65.0f);
    float pyc  = fminf(fmaxf(py, 0.0f), 65.0f);
    float glt = (1.0f + (qltx - pxc)) * (1.0f + (qlty - pyc));
    float grb = (1.0f - (qrbx - pxc)) * (1.0f - (qrby - pyc));
    float glb = (1.0f + (qltx - pxc)) * (1.0f - (qrby - pyc));
    float grt = (1.0f - (qrbx - pxc)) * (1.0f + (qlty - pyc));
    int ltx = (int)qltx, lty = (int)qlty, rbx = (int)qrbx, rby = (int)qrby;

    const float* plt = (ltx >= 1 && ltx <= 64 && lty >= 1 && lty <= 64)
        ? x + (((size_t)b * 64 + (ltx - 1)) * 64 + (lty - 1)) * 128 : nullptr;
    const float* prb = (rbx >= 1 && rbx <= 64 && rby >= 1 && rby <= 64)
        ? x + (((size_t)b * 64 + (rbx - 1)) * 64 + (rby - 1)) * 128 : nullptr;
    const float* plb = (ltx >= 1 && ltx <= 64 && rby >= 1 && rby <= 64)
        ? x + (((size_t)b * 64 + (ltx - 1)) * 64 + (rby - 1)) * 128 : nullptr;
    const float* prt = (rbx >= 1 && rbx <= 64 && lty >= 1 && lty <= 64)
        ? x + (((size_t)b * 64 + (rbx - 1)) * 64 + (lty - 1)) * 128 : nullptr;

    bf16* op = xo + (((size_t)b * 192 + (h * 3 + ix)) * 192 + (w * 3 + iy)) * 128;
#pragma unroll 4
    for (int c = 0; c < 128; ++c) {
        float v = 0.0f;
        if (plt) v += glt * plt[c];
        if (prb) v += grb * prb[c];
        if (plb) v += glb * plb[c];
        if (prt) v += grt * prt[c];
        op[c] = (bf16)v;
    }
}

// ---------------------------------------------------------------------------
// y = k_out * m + bias  (NHWC internal -> NCHW f32 d_out)
// ---------------------------------------------------------------------------
__global__ void combine_kernel(const float* __restrict__ k_out,
                               const float* __restrict__ m,
                               const float* __restrict__ bias,
                               float* __restrict__ out)
{
    int idx = blockIdx.x * blockDim.x + threadIdx.x;    // NCHW dest index
    if (idx >= 8 * 128 * 64 * 64) return;
    int w = idx & 63;
    int h = (idx >> 6) & 63;
    int c = (idx >> 12) & 127;
    int b = idx >> 19;
    size_t nhwc = (((size_t)b * 64 + h) * 64 + w) * 128 + c;
    out[idx] = k_out[nhwc] * m[nhwc] + bias[nhwc];
}

// ---------------------------------------------------------------------------
extern "C" void kernel_launch(void* const* d_in, const int* in_sizes, int n_in,
                              void* d_out, int out_size, void* d_ws, size_t ws_size,
                              hipStream_t stream)
{
    (void)in_sizes; (void)n_in; (void)out_size; (void)ws_size;
    const float* x     = (const float*)d_in[0];
    const float* m_w1  = (const float*)d_in[1];  const float* m_b1  = (const float*)d_in[2];
    const float* m_w2  = (const float*)d_in[3];  const float* m_b2  = (const float*)d_in[4];
    const float* m_w3  = (const float*)d_in[5];  const float* m_b3  = (const float*)d_in[6];
    const float* bb_w1 = (const float*)d_in[7];  const float* bb_b1 = (const float*)d_in[8];
    const float* bb_w2 = (const float*)d_in[9];  const float* bb_b2 = (const float*)d_in[10];
    const float* bb_w3 = (const float*)d_in[11]; const float* bb_b3 = (const float*)d_in[12];
    const float* p_w1  = (const float*)d_in[13]; const float* p_b1  = (const float*)d_in[14];
    const float* p_g1  = (const float*)d_in[15]; const float* p_be1 = (const float*)d_in[16];
    const float* p_w2  = (const float*)d_in[17]; const float* p_b2  = (const float*)d_in[18];
    const float* p_g2  = (const float*)d_in[19]; const float* p_be2 = (const float*)d_in[20];
    const float* l_w1  = (const float*)d_in[21]; const float* l_b1  = (const float*)d_in[22];
    const float* l_g1  = (const float*)d_in[23]; const float* l_be1 = (const float*)d_in[24];
    const float* l_w2  = (const float*)d_in[25]; const float* l_b2  = (const float*)d_in[26];
    const float* l_g2  = (const float*)d_in[27]; const float* l_be2 = (const float*)d_in[28];
    const float* w_w1  = (const float*)d_in[29]; const float* w_b1  = (const float*)d_in[30];
    const float* w_g1  = (const float*)d_in[31]; const float* w_be1 = (const float*)d_in[32];
    const float* w_w2  = (const float*)d_in[33]; const float* w_b2  = (const float*)d_in[34];
    const float* w_g2  = (const float*)d_in[35]; const float* w_be2 = (const float*)d_in[36];
    const float* k_w   = (const float*)d_in[37]; const float* k_b   = (const float*)d_in[38];
    const int*   hw_range = (const int*)d_in[40];

    // ---- workspace layout (256B aligned slabs) ----
    char*  ws  = (char*)d_ws;
    size_t woff = 0;
    auto alloc = [&](size_t bytes) -> void* {
        void* p = ws + woff;
        woff += (bytes + 255) & ~(size_t)255;
        return p;
    };
    const size_t ACT_F32 = (size_t)8 * 64 * 64 * 128 * 4;   // 16.78 MB
    const size_t ACT_BF  = (size_t)8 * 64 * 64 * 128 * 2;   //  8.39 MB
    float* x_f32    = (float*)alloc(ACT_F32);
    bf16*  x_bf     = (bf16*) alloc(ACT_BF);
    bf16*  x100_bf  = (bf16*) alloc(ACT_BF);
    bf16*  ping     = (bf16*) alloc(ACT_BF);
    bf16*  pong     = (bf16*) alloc(ACT_BF);
    float* m_f32    = (float*)alloc(ACT_F32);
    float* bias_f32 = (float*)alloc(ACT_F32);
    float* off_f32  = (float*)alloc(ACT_F32);
    float* lmap     = (float*)alloc((size_t)8 * 64 * 64 * 4);
    float* wmap     = (float*)alloc((size_t)8 * 64 * 64 * 4);
    bf16*  xo       = (bf16*) alloc((size_t)8 * 192 * 192 * 128 * 2);  // 75.5 MB
    bf16*  bpacks   = (bf16*) alloc((size_t)9 * 147456 * 2);
    float* k_f32    = x_f32;   // x no longer needed after sampling -> reuse

    const int ELT = 8 * 64 * 64 * 128;

    // 1) layout conversion
    prep_x_kernel<<<(ELT + 255) / 256, 256, 0, stream>>>(x, x_f32, x_bf, x100_bf);

    // 2) pack all 9 big-conv weights into WMMA B-fragment layout
    const float* ws_ptrs[9] = { m_w1, m_w2, m_w3, bb_w1, bb_w2, bb_w3, p_w1, p_w2, k_w };
    for (int i = 0; i < 9; ++i)
        pack_w_kernel<<<(147456 + 255) / 256, 256, 0, stream>>>(
            ws_ptrs[i], bpacks + (size_t)i * 147456);
    bf16* Bm1  = bpacks + (size_t)0 * 147456;
    bf16* Bm2  = bpacks + (size_t)1 * 147456;
    bf16* Bm3  = bpacks + (size_t)2 * 147456;
    bf16* Bbb1 = bpacks + (size_t)3 * 147456;
    bf16* Bbb2 = bpacks + (size_t)4 * 147456;
    bf16* Bbb3 = bpacks + (size_t)5 * 147456;
    bf16* Bp1  = bpacks + (size_t)6 * 147456;
    bf16* Bp2  = bpacks + (size_t)7 * 147456;
    bf16* Bk   = bpacks + (size_t)8 * 147456;

    dim3 cgrid(512), cblk(256);   // 512 blocks x 64 pixels = 32768 outputs
    // 3) m branch: lrelu, lrelu, tanh
    conv_wmma_t<1,1,64,64,1,false,false,true><<<cgrid, cblk, 0, stream>>>(
        x_bf, Bm1, m_b1, nullptr, nullptr, nullptr, ping);
    conv_wmma_t<1,1,64,64,1,false,false,true><<<cgrid, cblk, 0, stream>>>(
        ping, Bm2, m_b2, nullptr, nullptr, nullptr, pong);
    conv_wmma_t<1,1,64,64,2,false,true,false><<<cgrid, cblk, 0, stream>>>(
        pong, Bm3, m_b3, nullptr, nullptr, m_f32, nullptr);
    // bias branch: lrelu, lrelu, none
    conv_wmma_t<1,1,64,64,1,false,false,true><<<cgrid, cblk, 0, stream>>>(
        x_bf, Bbb1, bb_b1, nullptr, nullptr, nullptr, ping);
    conv_wmma_t<1,1,64,64,1,false,false,true><<<cgrid, cblk, 0, stream>>>(
        ping, Bbb2, bb_b2, nullptr, nullptr, nullptr, pong);
    conv_wmma_t<1,1,64,64,0,false,true,false><<<cgrid, cblk, 0, stream>>>(
        pong, Bbb3, bb_b3, nullptr, nullptr, bias_f32, nullptr);
    // p branch on x*100: (BN + lrelu) x2
    conv_wmma_t<1,1,64,64,1,true,false,true><<<cgrid, cblk, 0, stream>>>(
        x100_bf, Bp1, p_b1, p_g1, p_be1, nullptr, ping);
    conv_wmma_t<1,1,64,64,1,true,true,false><<<cgrid, cblk, 0, stream>>>(
        ping, Bp2, p_b2, p_g2, p_be2, off_f32, nullptr);

    // 4) l / w maps
    small_branch_kernel<<<(8 * 64 * 64 + 255) / 256, 256, 0, stream>>>(
        off_f32, l_w1, l_b1, l_g1, l_be1, l_w2, l_b2, l_g2, l_be2, hw_range, lmap);
    small_branch_kernel<<<(8 * 64 * 64 + 255) / 256, 256, 0, stream>>>(
        off_f32, w_w1, w_b1, w_g1, w_be1, w_w2, w_b2, w_g2, w_be2, hw_range, wmap);

    // 5) deformable bilinear sampling -> xo [8][192][192][128] bf16
    sample_kernel<<<(8 * 64 * 64 * 9 + 255) / 256, 256, 0, stream>>>(
        x_f32, lmap, wmap, xo);

    // 6) final 3x3 conv, stride 3, pad 0 over the expanded image
    conv_wmma_t<3,0,192,192,0,false,true,false><<<cgrid, cblk, 0, stream>>>(
        xo, Bk, k_b, nullptr, nullptr, k_f32, nullptr);

    // 7) y = k * m + bias, NCHW f32 output
    combine_kernel<<<(ELT + 255) / 256, 256, 0, stream>>>(
        k_f32, m_f32, bias_f32, (float*)d_out);
}
